// DeepseekMoEBlock_29076928594527
// MI455X (gfx1250) — matrix operand order repaired and checked by reading, not compiled, not run on hardware
//
#include <hip/hip_runtime.h>

// ---------------------------------------------------------------------------
// DeepSeek-style MoE block for MI455X (gfx1250), bf16 WMMA path.
//   T=2048 tokens, D=2048 model dim, F=1408 expert ffn, E=32 experts,
//   top-6 routing, capacity C=768, shared expert FS=2816.
//
// Roofline: fp32 weights dominate traffic (~1.13 GB -> ~48us @ 23.3 TB/s);
// 283 GFLOP of GEMM work needs matrix cores at bf16 (16x16x32) to stay
// balanced, so weights/activations are converted to bf16 in-flight and
// accumulated in fp32 via v_wmma_f32_16x16x32_bf16. GEMM K-loops are
// double-buffered in LDS so weight-stream latency hides under WMMA.
// ---------------------------------------------------------------------------

#define BM 64
#define BN 64
#define BK 32
#define LDSS 40          // ushort stride per LDS row (32 + 8 pad)

typedef __attribute__((ext_vector_type(16))) __bf16 bf16x16;
typedef __attribute__((ext_vector_type(2)))  __bf16 bf16x2;
typedef __attribute__((ext_vector_type(8)))  float  f32x8;

union FragAB { bf16x16 bf; unsigned int w[8]; };
union FragC  { f32x8 f; float e[8]; };

__device__ __forceinline__ unsigned int pack_bf16x2(float a, float b) {
  bf16x2 p; p.x = (__bf16)a; p.y = (__bf16)b;   // v_cvt_pk_bf16_f32
  return __builtin_bit_cast(unsigned int, p);
}
__device__ __forceinline__ unsigned short to_bf16(float a) {
  __bf16 h = (__bf16)a;
  return __builtin_bit_cast(unsigned short, h);
}
__device__ __forceinline__ float silu(float g) {
  return g * (1.0f / (1.0f + __expf(-g)));
}

// ---------------------------------------------------------------------------
// fp32 -> bf16 elementwise convert (vectorized x4)
// ---------------------------------------------------------------------------
__global__ __launch_bounds__(256) void cvt_bf16_kernel(
    const float* __restrict__ in, unsigned short* __restrict__ out, int n) {
  int i = (blockIdx.x * blockDim.x + threadIdx.x) * 4;
  if (i + 3 >= n) return;
  float4 v = *(const float4*)(in + i);
  uint2 r;
  r.x = pack_bf16x2(v.x, v.y);
  r.y = pack_bf16x2(v.z, v.w);
  *(uint2*)(out + i) = r;
}

// ---------------------------------------------------------------------------
// Router: one wave32 per token. Lane e computes logit for expert e (E==32),
// wave softmax, 6x wave argmax, atomic slot assignment.
// ---------------------------------------------------------------------------
__global__ __launch_bounds__(256) void router_kernel(
    const float* __restrict__ X, const float* __restrict__ Wr,
    int* __restrict__ cnt, int* __restrict__ rowlist, float* __restrict__ wlist,
    int T, int D, int C) {
  int lane = threadIdx.x & 31;
  int wave = threadIdx.x >> 5;
  int t = blockIdx.x * (blockDim.x >> 5) + wave;
  if (t >= T) return;

  const float4* xr = (const float4*)(X + (size_t)t * D);
  const float4* wr = (const float4*)(Wr + (size_t)lane * D);
  float acc = 0.f;
  int D4 = D >> 2;
  for (int i = 0; i < D4; ++i) {
    float4 a = xr[i], b = wr[i];
    acc += a.x * b.x + a.y * b.y + a.z * b.z + a.w * b.w;
  }
  float m = acc;
  for (int off = 16; off; off >>= 1) m = fmaxf(m, __shfl_xor(m, off));
  float p = __expf(acc - m);
  float s = p;
  for (int off = 16; off; off >>= 1) s += __shfl_xor(s, off);
  float score = p / s;

  float v = score;
  float wsel[6]; int isel[6];
  for (int k = 0; k < 6; ++k) {
    float mv = v; int mi = lane;
    for (int off = 16; off; off >>= 1) {
      float ov = __shfl_xor(mv, off);
      int   oi = __shfl_xor(mi, off);
      if (ov > mv || (ov == mv && oi < mi)) { mv = ov; mi = oi; }
    }
    wsel[k] = mv; isel[k] = mi;
    if (lane == mi) v = -1e30f;
  }
  float sum = 1e-20f;
  for (int k = 0; k < 6; ++k) sum += wsel[k];
  if (lane == 0) {
    for (int k = 0; k < 6; ++k) {
      int e = isel[k];
      int slot = atomicAdd(&cnt[e], 1);
      if (slot < C) {
        rowlist[e * C + slot] = t;
        wlist[e * C + slot] = wsel[k] / sum;
      }
    }
  }
}

// ---------------------------------------------------------------------------
// Fragment loaders (wave32, v_wmma_f32_16x16x32_bf16 layouts per ISA 7.12.2).
// Contiguous dword groups so the compiler merges them into ds_load_b128.
// ---------------------------------------------------------------------------
__device__ __forceinline__ void load_fragA(FragAB& a, const unsigned short* As,
                                           int rbase, int lane) {
  int half = lane >> 4;
  const unsigned short* p = As + (rbase + (lane & 15)) * LDSS;
#pragma unroll
  for (int j = 0; j < 4; ++j) {
    a.w[j]     = *(const unsigned int*)(p + half * 8 + 2 * j);
    a.w[j + 4] = *(const unsigned int*)(p + 16 + half * 8 + 2 * j);
  }
}
__device__ __forceinline__ void load_fragB(FragAB& b, const unsigned short* Bs,
                                           int cbase, int lane) {
  int half = lane >> 4;
  const unsigned short* p = Bs + (cbase + (lane & 15)) * LDSS;
#pragma unroll
  for (int j = 0; j < 4; ++j) {
    b.w[j]     = *(const unsigned int*)(p + half * 8 + 2 * j);
    b.w[j + 4] = *(const unsigned int*)(p + 16 + half * 8 + 2 * j);
  }
}

// Stage 4 consecutive-N fp32 values as bf16 into column-major LDS tile.
__device__ __forceinline__ void stage_b4(unsigned short* Bs, int f0, int dd,
                                         float4 v) {
  unsigned int u01 = pack_bf16x2(v.x, v.y);
  unsigned int u23 = pack_bf16x2(v.z, v.w);
  Bs[(f0 + 0) * LDSS + dd] = (unsigned short)(u01 & 0xFFFFu);
  Bs[(f0 + 1) * LDSS + dd] = (unsigned short)(u01 >> 16);
  Bs[(f0 + 2) * LDSS + dd] = (unsigned short)(u23 & 0xFFFFu);
  Bs[(f0 + 3) * LDSS + dd] = (unsigned short)(u23 >> 16);
}

// ---------------------------------------------------------------------------
// GEMM1 + SwiGLU: H[m,n] = silu(A@Wg)[m,n] * (A@Wu)[m,n]   (bf16 out)
//   A: gathered bf16 rows of x (rowmap) or dense (rowmap==nullptr).
//   Wg/Wu: fp32, converted to bf16 on the fly. K = D (2048).
//   Double-buffered LDS: 1 barrier/iter; next tile's global loads issue
//   before the WMMA block so their latency hides under matrix work.
// ---------------------------------------------------------------------------
__global__ __launch_bounds__(256) void gemm1_swiglu_kernel(
    const unsigned short* __restrict__ Xb, const float* __restrict__ Wg,
    const float* __restrict__ Wu, unsigned short* __restrict__ H,
    const int* __restrict__ rowmap, const int* __restrict__ cnt,
    int K, int Ncols, int tiles_per_e, int C) {
  __shared__ unsigned short As[2][BM * LDSS];
  __shared__ unsigned short Bgs[2][BN * LDSS];
  __shared__ unsigned short Bus[2][BN * LDSS];
  __shared__ int rowIds[BM];

  int n0 = blockIdx.x * BN;
  int m0, count, rowbase = 0;
  const float *wg = Wg, *wu = Wu;
  if (rowmap) {
    int e = blockIdx.y / tiles_per_e;
    m0 = (blockIdx.y % tiles_per_e) * BM;
    count = cnt[e]; if (count > C) count = C;
    if (m0 >= count) return;
    size_t wo = (size_t)e * K * Ncols;
    wg = Wg + wo; wu = Wu + wo;
    rowbase = e * C;
  } else {
    m0 = blockIdx.y * BM;
    count = gridDim.y * BM;
  }

  int tid = threadIdx.x;
  if (tid < BM) {
    int m = m0 + tid;
    rowIds[tid] = (m < count) ? (rowmap ? rowmap[rowbase + m] : m) : -1;
  }
  __syncthreads();

  int lane = tid & 31, wid = tid >> 5;
  int wm = wid >> 2;   // 0..1 -> 32-row strip
  int wn = wid & 3;    // 0..3 -> 16-col strip

  FragC cg0, cg1, cu0, cu1;
#pragma unroll
  for (int i = 0; i < 8; ++i) { cg0.e[i] = 0.f; cg1.e[i] = 0.f; cu0.e[i] = 0.f; cu1.e[i] = 0.f; }

  // ---- loop-invariant staging setup ----
  int ar = tid >> 2;                 // A tile row this thread stages
  int ak = (tid & 3) * 8;            // 8 bf16 (16B) per thread
  int myrow = rowIds[ar];            // constant across K loop
  const uint4* ap =
      (const uint4*)(Xb + (size_t)(myrow < 0 ? 0 : myrow) * K + ak);

  int f0 = (tid * 4) & 63;           // 4 consecutive N per float4
  int d0 = tid >> 4;                 // 0..15 (K row, +16 on 2nd pass)
  size_t s4 = (size_t)(Ncols >> 2);  // float4s per weight row
  const float4* wg4 = (const float4*)(wg + (size_t)d0 * Ncols + n0 + f0);
  const float4* wu4 = (const float4*)(wu + (size_t)d0 * Ncols + n0 + f0);

  // ---- prologue: fetch + stage tile 0 ----
  uint4 av = make_uint4(0u, 0u, 0u, 0u);
  if (myrow >= 0) av = ap[0];
  float4 vg0 = wg4[0], vg1 = wg4[16 * s4];
  float4 vu0 = wu4[0], vu1 = wu4[16 * s4];
  *(uint4*)(&As[0][ar * LDSS + ak]) = av;
  stage_b4(&Bgs[0][0], f0, d0, vg0);
  stage_b4(&Bgs[0][0], f0, d0 + 16, vg1);
  stage_b4(&Bus[0][0], f0, d0, vu0);
  stage_b4(&Bus[0][0], f0, d0 + 16, vu1);

  int nk = K / BK;
  for (int kt = 0; kt < nk; ++kt) {
    int buf = kt & 1;
    __syncthreads();
    bool more = (kt + 1) < nk;
    if (more) {
      // issue next tile's global loads (no wait yet)
      int k0 = (kt + 1) * BK;
      av = make_uint4(0u, 0u, 0u, 0u);
      if (myrow >= 0) av = ap[k0 >> 3];
      size_t ko = (size_t)k0 * s4;
      vg0 = wg4[ko]; vg1 = wg4[ko + 16 * s4];
      vu0 = wu4[ko]; vu1 = wu4[ko + 16 * s4];
      if (kt + 2 < nk) {
        __builtin_prefetch(wg4 + ko + (size_t)BK * s4, 0, 1);
        __builtin_prefetch(wu4 + ko + (size_t)BK * s4, 0, 1);
      }
    }

    FragAB a0, a1, bg, bu;
    load_fragA(a0, &As[buf][0], wm * 32, lane);
    load_fragA(a1, &As[buf][0], wm * 32 + 16, lane);
    load_fragB(bg, &Bgs[buf][0], wn * 16, lane);
    load_fragB(bu, &Bus[buf][0], wn * 16, lane);
    cg0.f = __builtin_amdgcn_wmma_f32_16x16x32_bf16(false, a0.bf, false, bg.bf, (short)0, cg0.f, false, false);
    cg1.f = __builtin_amdgcn_wmma_f32_16x16x32_bf16(false, a1.bf, false, bg.bf, (short)0, cg1.f, false, false);
    cu0.f = __builtin_amdgcn_wmma_f32_16x16x32_bf16(false, a0.bf, false, bu.bf, (short)0, cu0.f, false, false);
    cu1.f = __builtin_amdgcn_wmma_f32_16x16x32_bf16(false, a1.bf, false, bu.bf, (short)0, cu1.f, false, false);

    if (more) {
      int nb = buf ^ 1;
      *(uint4*)(&As[nb][ar * LDSS + ak]) = av;
      stage_b4(&Bgs[nb][0], f0, d0, vg0);
      stage_b4(&Bgs[nb][0], f0, d0 + 16, vg1);
      stage_b4(&Bus[nb][0], f0, d0, vu0);
      stage_b4(&Bus[nb][0], f0, d0 + 16, vu1);
    }
  }

  // ---- epilogue: silu(g)*u -> bf16 H ----
  int half = lane >> 4;
  int ncol = n0 + wn * 16 + (lane & 15);
#pragma unroll
  for (int i = 0; i < 8; ++i) {
    int m = m0 + wm * 32 + i + 8 * half;
    if (m < count)
      H[(size_t)(rowbase + m) * Ncols + ncol] = to_bf16(silu(cg0.e[i]) * cu0.e[i]);
    int m2 = m + 16;
    if (m2 < count)
      H[(size_t)(rowbase + m2) * Ncols + ncol] = to_bf16(silu(cg1.e[i]) * cu1.e[i]);
  }
}

// ---------------------------------------------------------------------------
// GEMM2 + weighted combine: Out[tok, n] += w * (H @ Wd)[m, n]
//   rowmap/wts==nullptr => dense (token=m, weight=1).  K = F or FS.
// ---------------------------------------------------------------------------
__global__ __launch_bounds__(256) void gemm2_combine_kernel(
    const unsigned short* __restrict__ Hm, const float* __restrict__ Wd,
    float* __restrict__ Out, const int* __restrict__ rowmap,
    const float* __restrict__ wts, const int* __restrict__ cnt,
    int K, int N, int tiles_per_e, int C) {
  __shared__ unsigned short As[2][BM * LDSS];
  __shared__ unsigned short Bs[2][BN * LDSS];
  __shared__ int rowTok[BM];
  __shared__ float rowW[BM];

  int n0 = blockIdx.x * BN;
  int m0, count, rowbase = 0;
  const float* wd = Wd;
  if (rowmap) {
    int e = blockIdx.y / tiles_per_e;
    m0 = (blockIdx.y % tiles_per_e) * BM;
    count = cnt[e]; if (count > C) count = C;
    if (m0 >= count) return;
    wd = Wd + (size_t)e * K * N;
    rowbase = e * C;
  } else {
    m0 = blockIdx.y * BM;
    count = gridDim.y * BM;
  }

  int tid = threadIdx.x;
  if (tid < BM) {
    int m = m0 + tid;
    if (m < count) {
      rowTok[tid] = rowmap ? rowmap[rowbase + m] : m;
      rowW[tid]   = wts ? wts[rowbase + m] : 1.0f;
    } else {
      rowTok[tid] = -1; rowW[tid] = 0.f;
    }
  }
  __syncthreads();

  int lane = tid & 31, wid = tid >> 5;
  int wm = wid >> 2, wn = wid & 3;

  FragC c0, c1;
#pragma unroll
  for (int i = 0; i < 8; ++i) { c0.e[i] = 0.f; c1.e[i] = 0.f; }

  int ar = tid >> 2;
  int ak = (tid & 3) * 8;
  bool avalid = (m0 + ar) < count;
  const uint4* ap =
      (const uint4*)(Hm + (size_t)(rowbase + (avalid ? m0 + ar : 0)) * K + ak);

  int f0 = (tid * 4) & 63;
  int d0 = tid >> 4;
  size_t s4 = (size_t)(N >> 2);
  const float4* wd4 = (const float4*)(wd + (size_t)d0 * N + n0 + f0);

  // ---- prologue: tile 0 ----
  uint4 av = make_uint4(0u, 0u, 0u, 0u);
  if (avalid) av = ap[0];
  float4 vd0 = wd4[0], vd1 = wd4[16 * s4];
  *(uint4*)(&As[0][ar * LDSS + ak]) = av;
  stage_b4(&Bs[0][0], f0, d0, vd0);
  stage_b4(&Bs[0][0], f0, d0 + 16, vd1);

  int nk = K / BK;
  for (int kt = 0; kt < nk; ++kt) {
    int buf = kt & 1;
    __syncthreads();
    bool more = (kt + 1) < nk;
    if (more) {
      int k0 = (kt + 1) * BK;
      av = make_uint4(0u, 0u, 0u, 0u);
      if (avalid) av = ap[k0 >> 3];
      size_t ko = (size_t)k0 * s4;
      vd0 = wd4[ko]; vd1 = wd4[ko + 16 * s4];
      if (kt + 2 < nk)
        __builtin_prefetch(wd4 + ko + (size_t)BK * s4, 0, 1);
    }

    FragAB a0, a1, b;
    load_fragA(a0, &As[buf][0], wm * 32, lane);
    load_fragA(a1, &As[buf][0], wm * 32 + 16, lane);
    load_fragB(b, &Bs[buf][0], wn * 16, lane);
    c0.f = __builtin_amdgcn_wmma_f32_16x16x32_bf16(false, a0.bf, false, b.bf, (short)0, c0.f, false, false);
    c1.f = __builtin_amdgcn_wmma_f32_16x16x32_bf16(false, a1.bf, false, b.bf, (short)0, c1.f, false, false);

    if (more) {
      int nb = buf ^ 1;
      *(uint4*)(&As[nb][ar * LDSS + ak]) = av;
      stage_b4(&Bs[nb][0], f0, d0, vd0);
      stage_b4(&Bs[nb][0], f0, d0 + 16, vd1);
    }
  }

  int half = lane >> 4;
  int ncol = n0 + wn * 16 + (lane & 15);
#pragma unroll
  for (int i = 0; i < 8; ++i) {
    int ml0 = wm * 32 + i + 8 * half;
    int t0 = rowTok[ml0];
    if (t0 >= 0) atomicAdd(&Out[(size_t)t0 * N + ncol], rowW[ml0] * c0.e[i]);
    int ml1 = ml0 + 16;
    int t1 = rowTok[ml1];
    if (t1 >= 0) atomicAdd(&Out[(size_t)t1 * N + ncol], rowW[ml1] * c1.e[i]);
  }
}

// ---------------------------------------------------------------------------
// Host-side orchestration
// ---------------------------------------------------------------------------
extern "C" void kernel_launch(void* const* d_in, const int* in_sizes, int n_in,
                              void* d_out, int out_size, void* d_ws, size_t ws_size,
                              hipStream_t stream) {
  (void)in_sizes; (void)n_in; (void)ws_size;
  constexpr int T = 2048, D = 2048, F = 1408, E = 32, FS = 2816;
  constexpr int TOPK = 6;
  constexpr int C = 2 * (T * TOPK / E);   // 768

  const float* x         = (const float*)d_in[0];
  const float* wg_router = (const float*)d_in[1];
  const float* w_gate    = (const float*)d_in[2];
  const float* w_up      = (const float*)d_in[3];
  const float* w_down    = (const float*)d_in[4];
  const float* ws_gate   = (const float*)d_in[5];
  const float* ws_up     = (const float*)d_in[6];
  const float* ws_down   = (const float*)d_in[7];
  float* out = (float*)d_out;

  char* ws = (char*)d_ws;
  size_t off = 0;
  auto alloc = [&](size_t bytes) -> void* {
    void* p = ws + off;
    off += (bytes + 255) & ~(size_t)255;
    return p;
  };
  unsigned short* x_bf = (unsigned short*)alloc((size_t)T * D * 2);
  unsigned short* h_e  = (unsigned short*)alloc((size_t)E * C * F * 2);
  unsigned short* h_s  = (unsigned short*)alloc((size_t)T * FS * 2);
  int*   cnt     = (int*)alloc(E * sizeof(int));
  int*   rowlist = (int*)alloc((size_t)E * C * sizeof(int));
  float* wlist   = (float*)alloc((size_t)E * C * sizeof(float));

  hipMemsetAsync(out, 0, (size_t)out_size * sizeof(float), stream);
  hipMemsetAsync(cnt, 0, E * sizeof(int), stream);

  cvt_bf16_kernel<<<(T * D) / 1024, 256, 0, stream>>>(x, x_bf, T * D);
  router_kernel<<<T / 8, 256, 0, stream>>>(x, wg_router, cnt, rowlist, wlist, T, D, C);

  // Expert MLPs (gathered rows, bf16 WMMA)
  gemm1_swiglu_kernel<<<dim3(F / BN, E * (C / BM)), 256, 0, stream>>>(
      x_bf, w_gate, w_up, h_e, rowlist, cnt, D, F, C / BM, C);
  gemm2_combine_kernel<<<dim3(D / BN, E * (C / BM)), 256, 0, stream>>>(
      h_e, w_down, out, rowlist, wlist, cnt, F, D, C / BM, C);

  // Shared expert (dense)
  gemm1_swiglu_kernel<<<dim3(FS / BN, T / BM), 256, 0, stream>>>(
      x_bf, ws_gate, ws_up, h_s, nullptr, nullptr, D, FS, 0, 0);
  gemm2_combine_kernel<<<dim3(D / BN, T / BM), 256, 0, stream>>>(
      h_s, ws_down, out, nullptr, nullptr, nullptr, FS, D, 0, 0);
}
